// AdvancedTransformerBlock_49108656062989
// MI455X (gfx1250) — matrix operand order repaired and checked by reading, not compiled, not run on hardware
//
#include <hip/hip_runtime.h>
#include <hip/hip_fp16.h>

typedef __attribute__((ext_vector_type(16))) _Float16 v16h;
typedef __attribute__((ext_vector_type(8)))  _Float16 v8h;
typedef __attribute__((ext_vector_type(8)))  float    v8f;
typedef __attribute__((ext_vector_type(4)))  unsigned int u32x4;
typedef __attribute__((ext_vector_type(8)))  int i32x8;
typedef __attribute__((ext_vector_type(4)))  int i32x4;

#define TOK  4096   // B*S tokens
#define DIM  1024
#define DFF  4096
#define SEQ  2048
#define NBAT 2
#define NH   16
#define HDM  64
#define NE   8
#define NROW 8192   // TOK * TOPK

// ---------------------------------------------------------------- helpers
__device__ inline v8f vzero8() {
  v8f r;
#pragma unroll
  for (int i = 0; i < 8; ++i) r[i] = 0.0f;
  return r;
}

__device__ inline v8f wmma16(v16h a, v16h b, v8f c) {
  // D = A(16x32 f16) * B(32x16 f16) + C(f32)
  return __builtin_amdgcn_wmma_f32_16x16x32_f16(false, a, false, b, (short)0, c,
                                                false, false);
}

__device__ inline float gelu_exact(float v) {
  return 0.5f * v * (1.0f + erff(v * 0.70710678118654752f));
}

// ---- Tensor Data Mover: 2-D tile load, global -> LDS (CDNA5 TDM, D# per ISA §8)
// tile_x = contiguous elements per row, tile_y = rows, stride_elems = row pitch.
// pad_interval_code: +1 DWORD LDS padding after every 2^(code+1) DWORDs stored.
__device__ inline void tdm_load_2d(unsigned lds_off, const void* gptr,
                                   unsigned data_size_code, unsigned tile_x,
                                   unsigned tile_y,
                                   unsigned long long stride_elems,
                                   unsigned pad_interval_code) {
  unsigned long long ga = (unsigned long long)(uintptr_t)gptr;
  u32x4 g0;
  g0[0] = 1u;                                   // count=1 (valid user D#)
  g0[1] = lds_off;                              // lds_addr (bytes)
  g0[2] = (unsigned)ga;                         // global_addr[31:0]
  g0[3] = (unsigned)((ga >> 32) & 0x1FFFFFFu)   // global_addr[56:32]
          | (2u << 30);                         // type = 2 ("image")
  i32x8 g1;
  g1[0] = (int)((data_size_code << 16)          // 1=2B(f16), 2=4B(f32)
                | (1u << 20)                    // pad_enable
                | (pad_interval_code << 22));   // pad_amount field 0 => 1 DWORD
  g1[1] = (int)((tile_x & 0xFFFFu) << 16);      // tensor_dim0 = tile_x (no clip)
  g1[2] = (int)(((tile_x >> 16) & 0xFFFFu) | ((tile_y & 0xFFFFu) << 16));
  g1[3] = (int)(((tile_y >> 16) & 0xFFFFu) | ((tile_x & 0xFFFFu) << 16)); // tile_dim0
  g1[4] = (int)(tile_y & 0xFFFFu);              // tile_dim1 (tile_dim2=0 -> 2D)
  g1[5] = (int)(stride_elems & 0xFFFFFFFFu);    // tensor_dim0_stride[31:0]
  g1[6] = (int)((stride_elems >> 32) & 0xFFFFu);
  g1[7] = 0;
  i32x4 z4;
  z4[0] = z4[1] = z4[2] = z4[3] = 0;
  i32x8 z8;
#pragma unroll
  for (int i = 0; i < 8; ++i) z8[i] = 0;
  __builtin_amdgcn_tensor_load_to_lds(g0, g1, z4, z4, z8, 0);
}

// ---------------------------------------------------------------- LayerNorm
__global__ __launch_bounds__(256)
void ln_kernel(const float* __restrict__ x, const float* __restrict__ g,
               const float* __restrict__ bb, _Float16* __restrict__ out16,
               float* __restrict__ out32) {
  const int t = blockIdx.x;
  const int tid = threadIdx.x;
  const float* row = x + (size_t)t * DIM;
  __shared__ float red[256];

  float s = 0.0f;
#pragma unroll
  for (int i = tid; i < DIM; i += 256) s += row[i];
  red[tid] = s;
  __syncthreads();
  for (int st = 128; st > 0; st >>= 1) {
    if (tid < st) red[tid] += red[tid + st];
    __syncthreads();
  }
  const float mu = red[0] * (1.0f / DIM);
  __syncthreads();

  float s2 = 0.0f;
#pragma unroll
  for (int i = tid; i < DIM; i += 256) {
    float d = row[i] - mu;
    s2 += d * d;
  }
  red[tid] = s2;
  __syncthreads();
  for (int st = 128; st > 0; st >>= 1) {
    if (tid < st) red[tid] += red[tid + st];
    __syncthreads();
  }
  const float rs = rsqrtf(red[0] * (1.0f / DIM) + 1e-5f);

#pragma unroll
  for (int i = tid; i < DIM; i += 256) {
    float y = (row[i] - mu) * rs * g[i] + bb[i];
    out16[(size_t)t * DIM + i] = (_Float16)y;
    if (out32) out32[(size_t)t * DIM + i] = y;
  }
}

// ---------------------------------------------------------------- GEMM core
// Block: 256 thr = 8 waves (2 x 4). Tile 64(M) x 128(N), BK=32.
// A: f16 row-major [*,K]; Bw: f32 row-major [K,N].
// Tiles are staged via TDM (tensor_load_to_lds) with +1 DWORD row padding.
struct GemmAcc {
  v8f c[2][2];
  int wm, wn, half, lr;
};

template <bool MASK_ROWS>
__device__ inline GemmAcc gemm_core(const _Float16* __restrict__ A,
                                    const float* __restrict__ Bw, int K, int N,
                                    int row0, int rows_valid, int n0) {
  __shared__ _Float16 As[64 * 34 + 2];  // 64 rows, stride 34 halves (32+pad)
  __shared__ float Bsf[32 * 129];       // 32 rows, stride 129 floats (128+pad)
  const int tid = threadIdx.x;
  const int lane = tid & 31, w = tid >> 5;
  GemmAcc G;
  G.wm = w >> 2;
  G.wn = w & 3;
  G.half = lane >> 4;
  G.lr = lane & 15;
#pragma unroll
  for (int mt = 0; mt < 2; ++mt)
#pragma unroll
    for (int ct = 0; ct < 2; ++ct) G.c[mt][ct] = vzero8();

  const unsigned asOff = (unsigned)(uintptr_t)&As[0];
  const unsigned bsOff = (unsigned)(uintptr_t)&Bsf[0];
  const unsigned rowsA = MASK_ROWS ? (unsigned)rows_valid : 64u;

  for (int k0 = 0; k0 < K; k0 += 32) {
    __syncthreads();  // previous fragments consumed before TDM overwrites LDS
    if (tid < 32) {   // wave 0 issues the DMA descriptors
      // A tile: 64(rows) x 32(k) f16, row pitch K; 16 DW/row -> pad code 3
      tdm_load_2d(asOff, A + (size_t)row0 * K + k0, 1u, 32u, rowsA,
                  (unsigned long long)K, 3u);
      // B tile: 32(k rows) x 128(n) f32, row pitch N; 128 DW/row -> pad code 6
      tdm_load_2d(bsOff, Bw + (size_t)k0 * N + n0, 2u, 128u, 32u,
                  (unsigned long long)N, 6u);
      __builtin_amdgcn_s_wait_tensorcnt((short)0);
    }
    __syncthreads();  // publish LDS tile to all 8 waves

    // ---- fragments + 4 WMMA
    v16h af[2], bf[2];
#pragma unroll
    for (int mt = 0; mt < 2; ++mt) {
      const _Float16* ap2 = &As[(G.wm * 32 + mt * 16 + G.lr) * 34];
#pragma unroll
      for (int j = 0; j < 8; ++j) {
        af[mt][j] = ap2[G.half * 8 + j];
        af[mt][8 + j] = ap2[16 + G.half * 8 + j];
      }
    }
#pragma unroll
    for (int ct = 0; ct < 2; ++ct) {
      const int col = G.wn * 32 + ct * 16 + G.lr;
#pragma unroll
      for (int j = 0; j < 16; ++j)
        bf[ct][j] = (_Float16)Bsf[(G.half * 8 + (j & 7) + ((j >> 3) << 4)) * 129 + col];
    }
#pragma unroll
    for (int mt = 0; mt < 2; ++mt)
#pragma unroll
      for (int ct = 0; ct < 2; ++ct)
        G.c[mt][ct] = wmma16(af[mt], bf[ct], G.c[mt][ct]);
  }
  return G;
}

// ---------------------------------------------------------------- QKV GEMM
__global__ __launch_bounds__(256)
void gemm_qkv_kernel(const _Float16* __restrict__ h16,
                     const float* __restrict__ wqkv,
                     const float* __restrict__ bqkv, _Float16* __restrict__ q16,
                     _Float16* __restrict__ k16, _Float16* __restrict__ v16) {
  const int n0 = blockIdx.x * 128, row0 = blockIdx.y * 64;
  GemmAcc G = gemm_core<false>(h16, wqkv, DIM, 3 * DIM, row0, 64, n0);
#pragma unroll
  for (int mt = 0; mt < 2; ++mt)
#pragma unroll
    for (int ct = 0; ct < 2; ++ct)
#pragma unroll
      for (int v = 0; v < 8; ++v) {
        int row = row0 + G.wm * 32 + mt * 16 + v + 8 * G.half;  // token
        int col = n0 + G.wn * 32 + ct * 16 + G.lr;              // 0..3071
        float val = G.c[mt][ct][v] + bqkv[col];
        int part = col >> 10;  // 0=q 1=k 2=v
        int d = col & 1023;
        int hh = d >> 6, hd = d & 63;
        int b = row >> 11, s = row & 2047;
        size_t idx = (((size_t)b * NH + hh) * SEQ + s) * HDM + hd;
        _Float16 hv = (_Float16)val;
        if (part == 0)      q16[idx] = hv;
        else if (part == 1) k16[idx] = hv;
        else                v16[idx] = hv;
      }
}

// ---------------------------------------------------------------- attention
// one wave per (b,h, 16-query tile); flash-style online softmax with WMMA
__global__ __launch_bounds__(32)
void attn_kernel(const _Float16* __restrict__ q16,
                 const _Float16* __restrict__ k16,
                 const _Float16* __restrict__ v16, _Float16* __restrict__ ctx16) {
  const int tile = blockIdx.x;
  const int qt = tile & (SEQ / 16 - 1);  // 128 q-tiles per (b,h)
  const int bh = tile >> 7;              // b*NH + h
  const int lane = threadIdx.x;
  const int half = lane >> 4, lr = lane & 15;

  const _Float16* Q = q16 + ((size_t)bh * SEQ + qt * 16) * HDM;
  const _Float16* K = k16 + (size_t)bh * SEQ * HDM;
  const _Float16* V = v16 + (size_t)bh * SEQ * HDM;

  __shared__ _Float16 VT[64][40];  // [dim][key-in-chunk]
  __shared__ _Float16 P[16][40];   // [q-row][key-in-chunk]

  v16h a0, a1;
  {
    const _Float16* qp = Q + (size_t)lr * HDM;
#pragma unroll
    for (int j = 0; j < 8; ++j) {
      a0[j] = qp[half * 8 + j];
      a0[8 + j] = qp[16 + half * 8 + j];
      a1[j] = qp[32 + half * 8 + j];
      a1[8 + j] = qp[48 + half * 8 + j];
    }
  }
  v8f acc0 = vzero8(), acc1 = vzero8(), acc2 = vzero8(), acc3 = vzero8();
  float rowm[8], rowl[8];
#pragma unroll
  for (int v = 0; v < 8; ++v) { rowm[v] = -1e30f; rowl[v] = 0.0f; }

  const int nchunk = (qt * 16 + 47) >> 5;
  for (int ch = 0; ch < nchunk; ++ch) {
    const int kb = ch * 32;
    {
      const _Float16* vp = V + (size_t)(kb + lane) * HDM;
#pragma unroll
      for (int d0 = 0; d0 < 64; d0 += 8) {
        v8h vv = *(const v8h*)(vp + d0);
#pragma unroll
        for (int j = 0; j < 8; ++j) VT[d0 + j][lane] = vv[j];
      }
    }
    v8f c0 = vzero8(), c1 = vzero8();
    {
      const _Float16* kp0 = K + (size_t)(kb + lr) * HDM;
      const _Float16* kp1 = K + (size_t)(kb + 16 + lr) * HDM;
      v16h bk;
#pragma unroll
      for (int j = 0; j < 16; ++j) bk[j] = kp0[half * 8 + (j & 7) + ((j >> 3) << 4)];
      c0 = wmma16(a0, bk, c0);
#pragma unroll
      for (int j = 0; j < 16; ++j) bk[j] = kp0[32 + half * 8 + (j & 7) + ((j >> 3) << 4)];
      c0 = wmma16(a1, bk, c0);
#pragma unroll
      for (int j = 0; j < 16; ++j) bk[j] = kp1[half * 8 + (j & 7) + ((j >> 3) << 4)];
      c1 = wmma16(a0, bk, c1);
#pragma unroll
      for (int j = 0; j < 16; ++j) bk[j] = kp1[32 + half * 8 + (j & 7) + ((j >> 3) << 4)];
      c1 = wmma16(a1, bk, c1);
    }
    float p0[8], p1[8];
#pragma unroll
    for (int v = 0; v < 8; ++v) {
      const int qrow = qt * 16 + v + 8 * half;
      float s0 = c0[v] * 0.125f;
      float s1 = c1[v] * 0.125f;
      if (kb + lr > qrow) s0 = -1e30f;
      if (kb + 16 + lr > qrow) s1 = -1e30f;
      float mx = fmaxf(s0, s1);
#pragma unroll
      for (int off = 1; off < 16; off <<= 1) mx = fmaxf(mx, __shfl_xor(mx, off, 32));
      const float newm = fmaxf(rowm[v], mx);
      const float sc = expf(rowm[v] - newm);
      p0[v] = expf(s0 - newm);
      p1[v] = expf(s1 - newm);
      float sum = p0[v] + p1[v];
#pragma unroll
      for (int off = 1; off < 16; off <<= 1) sum += __shfl_xor(sum, off, 32);
      rowl[v] = rowl[v] * sc + sum;
      rowm[v] = newm;
      acc0[v] *= sc; acc1[v] *= sc; acc2[v] *= sc; acc3[v] *= sc;
    }
#pragma unroll
    for (int v = 0; v < 8; ++v) {
      P[v + 8 * half][lr] = (_Float16)p0[v];
      P[v + 8 * half][16 + lr] = (_Float16)p1[v];
    }
    __syncthreads();
    v16h pa;
    {
      const _Float16* pp = &P[lr][0];
#pragma unroll
      for (int j = 0; j < 8; ++j) {
        pa[j] = pp[half * 8 + j];
        pa[8 + j] = pp[16 + half * 8 + j];
      }
    }
#pragma unroll
    for (int nt = 0; nt < 4; ++nt) {
      v16h bv;
      const _Float16* vtp = &VT[nt * 16 + lr][0];
#pragma unroll
      for (int j = 0; j < 16; ++j) bv[j] = vtp[half * 8 + (j & 7) + ((j >> 3) << 4)];
      if (nt == 0)      acc0 = wmma16(pa, bv, acc0);
      else if (nt == 1) acc1 = wmma16(pa, bv, acc1);
      else if (nt == 2) acc2 = wmma16(pa, bv, acc2);
      else              acc3 = wmma16(pa, bv, acc3);
    }
    __syncthreads();
  }
  const int b = bh >> 4, hh = bh & 15;
#pragma unroll
  for (int v = 0; v < 8; ++v) {
    const int s = qt * 16 + v + 8 * half;
    const float inv = 1.0f / rowl[v];
    const size_t base = (((size_t)b * SEQ + s) * NH + hh) * HDM;
    ctx16[base + 0 + lr]  = (_Float16)(acc0[v] * inv);
    ctx16[base + 16 + lr] = (_Float16)(acc1[v] * inv);
    ctx16[base + 32 + lr] = (_Float16)(acc2[v] * inv);
    ctx16[base + 48 + lr] = (_Float16)(acc3[v] * inv);
  }
}

// ---------------------------------------------------------------- WO + residual
__global__ __launch_bounds__(256)
void gemm_wo_kernel(const _Float16* __restrict__ ctx16,
                    const float* __restrict__ wo, const float* __restrict__ bo,
                    const float* __restrict__ x, float* __restrict__ x2) {
  const int n0 = blockIdx.x * 128, row0 = blockIdx.y * 64;
  GemmAcc G = gemm_core<false>(ctx16, wo, DIM, DIM, row0, 64, n0);
#pragma unroll
  for (int mt = 0; mt < 2; ++mt)
#pragma unroll
    for (int ct = 0; ct < 2; ++ct)
#pragma unroll
      for (int v = 0; v < 8; ++v) {
        int row = row0 + G.wm * 32 + mt * 16 + v + 8 * G.half;
        int col = n0 + G.wn * 32 + ct * 16 + G.lr;
        size_t idx = (size_t)row * DIM + col;
        x2[idx] = x[idx] + G.c[mt][ct][v] + bo[col];
      }
}

// ---------------------------------------------------------------- router
__global__ __launch_bounds__(256)
void router_kernel(const float* __restrict__ h2, const float* __restrict__ wr,
                   int* __restrict__ top_i, float* __restrict__ top_g) {
  const int t = blockIdx.x * 256 + threadIdx.x;
  if (t >= TOK) return;
  const float* hp = h2 + (size_t)t * DIM;
  float lg[NE];
#pragma unroll
  for (int e = 0; e < NE; ++e) lg[e] = 0.0f;
  for (int d = 0; d < DIM; ++d) {
    const float hv = hp[d];
    const float* wp = wr + (size_t)d * NE;
#pragma unroll
    for (int e = 0; e < NE; ++e) lg[e] += hv * wp[e];
  }
  int i0 = 0;
#pragma unroll
  for (int e = 1; e < NE; ++e)
    if (lg[e] > lg[i0]) i0 = e;
  int i1 = (i0 == 0) ? 1 : 0;
#pragma unroll
  for (int e = 0; e < NE; ++e)
    if (e != i0 && lg[e] > lg[i1]) i1 = e;
  const float m = fmaxf(lg[i0], lg[i1]);
  const float e0 = expf(lg[i0] - m), e1 = expf(lg[i1] - m);
  const float inv = 1.0f / (e0 + e1);
  top_i[2 * t + 0] = i0;
  top_i[2 * t + 1] = i1;
  top_g[2 * t + 0] = e0 * inv;
  top_g[2 * t + 1] = e1 * inv;
}

// ---------------------------------------------------------------- MoE dispatch
__global__ void moe_count_kernel(const int* __restrict__ top_i,
                                 int* __restrict__ cnt) {
  const int t = blockIdx.x * 256 + threadIdx.x;
  if (t >= TOK) return;
  atomicAdd(&cnt[top_i[2 * t + 0]], 1);
  atomicAdd(&cnt[top_i[2 * t + 1]], 1);
}

__global__ void moe_scan_kernel(const int* __restrict__ cnt,
                                int* __restrict__ off, int* __restrict__ cur) {
  if (threadIdx.x == 0 && blockIdx.x == 0) {
    int acc = 0;
    for (int e = 0; e < NE; ++e) {
      off[e] = acc;
      cur[e] = acc;
      acc += cnt[e];
    }
  }
}

__global__ void moe_assign_kernel(const int* __restrict__ top_i,
                                  int* __restrict__ cur,
                                  int* __restrict__ tok2row) {
  const int t = blockIdx.x * 256 + threadIdx.x;
  if (t >= TOK) return;
#pragma unroll
  for (int k = 0; k < 2; ++k) {
    const int e = top_i[2 * t + k];
    const int pos = atomicAdd(&cur[e], 1);  // nondet position, det. row content
    tok2row[2 * t + k] = pos;
  }
}

__global__ __launch_bounds__(256)
void moe_gather_kernel(const int* __restrict__ tok2row,
                       const _Float16* __restrict__ h2_16,
                       _Float16* __restrict__ A_moe) {
  const int slot = blockIdx.x;  // 0..NROW-1
  const int t = slot >> 1;
  const int row = tok2row[slot];
  const uint2* src = (const uint2*)(h2_16 + (size_t)t * DIM);
  uint2* dst = (uint2*)(A_moe + (size_t)row * DIM);
  dst[threadIdx.x] = src[threadIdx.x];  // 256 * 8B = 2KB = DIM halves
}

// ---------------------------------------------------------------- MoE GEMMs
__global__ __launch_bounds__(256)
void gemm_moe1_kernel(const _Float16* __restrict__ A_moe,
                      const float* __restrict__ w1, const float* __restrict__ b1,
                      const int* __restrict__ off, const int* __restrict__ cnt,
                      _Float16* __restrict__ hid16) {
  const int e = blockIdx.z;
  const int cnt_e = cnt[e], off_e = off[e];
  const int y = blockIdx.y;
  if (y * 64 >= cnt_e) return;
  const int row0 = off_e + y * 64;
  const int rv = (cnt_e - y * 64 < 64) ? (cnt_e - y * 64) : 64;
  const int n0 = blockIdx.x * 128;
  GemmAcc G = gemm_core<true>(A_moe, w1 + (size_t)e * DIM * DFF, DIM, DFF,
                              row0, rv, n0);
#pragma unroll
  for (int mt = 0; mt < 2; ++mt)
#pragma unroll
    for (int ct = 0; ct < 2; ++ct)
#pragma unroll
      for (int v = 0; v < 8; ++v) {
        const int rl = G.wm * 32 + mt * 16 + v + 8 * G.half;
        if (rl >= rv) continue;
        const int col = n0 + G.wn * 32 + ct * 16 + G.lr;
        const float val = gelu_exact(G.c[mt][ct][v] + b1[(size_t)e * DFF + col]);
        hid16[(size_t)(row0 + rl) * DFF + col] = (_Float16)val;
      }
}

__global__ __launch_bounds__(256)
void gemm_moe2_kernel(const _Float16* __restrict__ hid16,
                      const float* __restrict__ w2, const float* __restrict__ b2,
                      const int* __restrict__ off, const int* __restrict__ cnt,
                      float* __restrict__ eout) {
  const int e = blockIdx.z;
  const int cnt_e = cnt[e], off_e = off[e];
  const int y = blockIdx.y;
  if (y * 64 >= cnt_e) return;
  const int row0 = off_e + y * 64;
  const int rv = (cnt_e - y * 64 < 64) ? (cnt_e - y * 64) : 64;
  const int n0 = blockIdx.x * 128;
  GemmAcc G = gemm_core<true>(hid16, w2 + (size_t)e * DFF * DIM, DFF, DIM,
                              row0, rv, n0);
#pragma unroll
  for (int mt = 0; mt < 2; ++mt)
#pragma unroll
    for (int ct = 0; ct < 2; ++ct)
#pragma unroll
      for (int v = 0; v < 8; ++v) {
        const int rl = G.wm * 32 + mt * 16 + v + 8 * G.half;
        if (rl >= rv) continue;
        const int col = n0 + G.wn * 32 + ct * 16 + G.lr;
        eout[(size_t)(row0 + rl) * DIM + col] =
            G.c[mt][ct][v] + b2[(size_t)e * DIM + col];
      }
}

// ---------------------------------------------------------------- combine
__global__ __launch_bounds__(256)
void combine_kernel(const float* __restrict__ x2, const float* __restrict__ eout,
                    const int* __restrict__ tok2row,
                    const float* __restrict__ top_g, float* __restrict__ out) {
  const int t = blockIdx.x;
  const int r0 = tok2row[2 * t + 0], r1 = tok2row[2 * t + 1];
  const float g0 = top_g[2 * t + 0], g1 = top_g[2 * t + 1];
  const float* e0 = eout + (size_t)r0 * DIM;
  const float* e1 = eout + (size_t)r1 * DIM;
#pragma unroll
  for (int i = threadIdx.x; i < DIM; i += 256)
    out[(size_t)t * DIM + i] = x2[(size_t)t * DIM + i] + g0 * e0[i] + g1 * e1[i];
}

// ---------------------------------------------------------------- launch
extern "C" void kernel_launch(void* const* d_in, const int* in_sizes, int n_in,
                              void* d_out, int out_size, void* d_ws,
                              size_t ws_size, hipStream_t stream) {
  (void)in_sizes; (void)n_in; (void)out_size; (void)ws_size;
  const float* x       = (const float*)d_in[0];
  const float* ln1_g   = (const float*)d_in[1];
  const float* ln1_b   = (const float*)d_in[2];
  const float* ln2_g   = (const float*)d_in[3];
  const float* ln2_b   = (const float*)d_in[4];
  const float* wqkv    = (const float*)d_in[5];
  const float* bqkv    = (const float*)d_in[6];
  const float* wo      = (const float*)d_in[7];
  const float* bo      = (const float*)d_in[8];
  const float* wrouter = (const float*)d_in[9];
  const float* w1      = (const float*)d_in[10];
  const float* b1      = (const float*)d_in[11];
  const float* w2      = (const float*)d_in[12];
  const float* b2      = (const float*)d_in[13];

  char* p = (char*)d_ws;
  auto alloc = [&](size_t bytes) -> void* {
    void* r = (void*)p;
    p += (bytes + 255) & ~(size_t)255;
    return r;
  };
  _Float16* h16    = (_Float16*)alloc((size_t)TOK * DIM * 2);
  _Float16* h2_16  = (_Float16*)alloc((size_t)TOK * DIM * 2);
  float*    h2f    = (float*)alloc((size_t)TOK * DIM * 4);
  _Float16* q16    = (_Float16*)alloc((size_t)NBAT * NH * SEQ * HDM * 2);
  _Float16* k16    = (_Float16*)alloc((size_t)NBAT * NH * SEQ * HDM * 2);
  _Float16* v16    = (_Float16*)alloc((size_t)NBAT * NH * SEQ * HDM * 2);
  _Float16* ctx16  = (_Float16*)alloc((size_t)TOK * DIM * 2);
  float*    x2     = (float*)alloc((size_t)TOK * DIM * 4);
  int*      top_i  = (int*)alloc((size_t)TOK * 2 * 4);
  float*    top_g  = (float*)alloc((size_t)TOK * 2 * 4);
  int*      cnt    = (int*)alloc(NE * 4);
  int*      off    = (int*)alloc(NE * 4);
  int*      cur    = (int*)alloc(NE * 4);
  int*      t2r    = (int*)alloc((size_t)TOK * 2 * 4);
  _Float16* A_moe  = (_Float16*)alloc((size_t)NROW * DIM * 2);
  _Float16* hid16  = (_Float16*)alloc((size_t)NROW * DFF * 2);
  float*    eout   = (float*)alloc((size_t)NROW * DIM * 4);

  (void)hipMemsetAsync(cnt, 0, NE * 4, stream);

  // attention half
  ln_kernel<<<TOK, 256, 0, stream>>>(x, ln1_g, ln1_b, h16, nullptr);
  gemm_qkv_kernel<<<dim3(3 * DIM / 128, TOK / 64), 256, 0, stream>>>(
      h16, wqkv, bqkv, q16, k16, v16);
  attn_kernel<<<NBAT * NH * (SEQ / 16), 32, 0, stream>>>(q16, k16, v16, ctx16);
  gemm_wo_kernel<<<dim3(DIM / 128, TOK / 64), 256, 0, stream>>>(ctx16, wo, bo,
                                                                x, x2);
  // MoE half
  ln_kernel<<<TOK, 256, 0, stream>>>(x2, ln2_g, ln2_b, h2_16, h2f);
  router_kernel<<<TOK / 256, 256, 0, stream>>>(h2f, wrouter, top_i, top_g);
  moe_count_kernel<<<TOK / 256, 256, 0, stream>>>(top_i, cnt);
  moe_scan_kernel<<<1, 32, 0, stream>>>(cnt, off, cur);
  moe_assign_kernel<<<TOK / 256, 256, 0, stream>>>(top_i, cur, t2r);
  moe_gather_kernel<<<NROW, 256, 0, stream>>>(t2r, h2_16, A_moe);
  gemm_moe1_kernel<<<dim3(DFF / 128, TOK / 64, NE), 256, 0, stream>>>(
      A_moe, w1, b1, off, cnt, hid16);
  gemm_moe2_kernel<<<dim3(DIM / 128, TOK / 64, NE), 256, 0, stream>>>(
      hid16, w2, b2, off, cnt, eout);
  combine_kernel<<<TOK, 256, 0, stream>>>(x2, eout, t2r, top_g, (float*)d_out);
}
